// MultiHeadAttention_52029233823859
// MI455X (gfx1250) — compile-verified, matrix operand order checked by reading, not run on hardware
//
#include <hip/hip_runtime.h>
#include <cstdint>

#define B_BATCH 2
#define S_LEN   2048
#define D_MODEL 1024
#define NH      16
#define DK      64

// GEMM tiling: block = 256 threads = 8 waves; block tile 128(M) x 64(N), K-step 32.
#define BM 128
#define BN 64
#define BK 32
#define LDT 40   // LDS tile row stride in halves (BK + 8 pad; 80B rows keep 16B alignment)

typedef __attribute__((ext_vector_type(16))) _Float16 v16h;
typedef __attribute__((ext_vector_type(8)))  float    v8f;
typedef int v4i __attribute__((vector_size(16)));   // matches builtin param type

union FragH { v16h v; _Float16 h[16]; };
union FragF { v8f  v; float    f[8];  };

#define AS1 __attribute__((address_space(1)))
#define AS3 __attribute__((address_space(3)))

#if defined(__has_builtin)
# if __has_builtin(__builtin_amdgcn_global_load_async_to_lds_b128)
#  define HAVE_ASYNC_LDS 1
# endif
#endif

// 16-byte global -> LDS copy. Uses the CDNA5 async-to-LDS path (ASYNCcnt)
// when the toolchain exposes it; otherwise a synchronous VGPR bounce.
__device__ __forceinline__ void copy16_g2l(const void* g, void* l) {
#ifdef HAVE_ASYNC_LDS
  __builtin_amdgcn_global_load_async_to_lds_b128(
      (AS1 v4i*)g, (AS3 v4i*)(uint32_t)(uintptr_t)l, 0, 0);
#else
  *(uint4*)l = *(const uint4*)g;
#endif
}

__device__ __forceinline__ void async_wait0() {
#ifdef HAVE_ASYNC_LDS
# if __has_builtin(__builtin_amdgcn_s_wait_asynccnt)
  __builtin_amdgcn_s_wait_asynccnt(0);
# else
  asm volatile("s_wait_asynccnt 0" ::: "memory");
# endif
#endif
}

// ---------------------------------------------------------------------------
// Fragment load from an LDS tile stored row-major with stride LDT.
// A (16x32, MxK): lane holds M = lane%16; K striping per ISA table
// (VGPR0..3: K = kb+0..7, VGPR4..7: K = 16+kb+0..7, kb = (lane<16)?0:8).
// B tiles are stored [N][K] in LDS so the same loader yields the B fragment.
// The half-pair loads fuse into ds_load_b128s.
// ---------------------------------------------------------------------------
__device__ __forceinline__ v16h lds_frag(const _Float16* base, int lane) {
  FragH f;
  const int r  = lane & 15;
  const int kb = (lane & 16) ? 8 : 0;
  const _Float16* row = base + r * LDT;
#pragma unroll
  for (int i = 0; i < 4; ++i) {
    f.h[2*i]         = row[kb + 2*i];
    f.h[2*i + 1]     = row[kb + 2*i + 1];
    f.h[8 + 2*i]     = row[16 + kb + 2*i];
    f.h[8 + 2*i + 1] = row[16 + kb + 2*i + 1];
  }
  return f.v;
}

__device__ __forceinline__ v8f wmma_step(v16h a, v16h b, v8f c) {
  return __builtin_amdgcn_wmma_f32_16x16x32_f16(
      /*neg_a=*/false, a, /*neg_b=*/false, b,
      /*c_mod=*/(short)0, c, /*reuse_a=*/false, /*reuse_b=*/false);
}

// ---------------------------------------------------------------------------
// fp32 -> f16 conversion (exact multiples of 1024 elements per launch here)
// ---------------------------------------------------------------------------
__global__ void __launch_bounds__(256)
cvt_f32_f16(const float* __restrict__ src, _Float16* __restrict__ dst, int n) {
  int i = (blockIdx.x * 256 + threadIdx.x) * 4;
  if (i + 3 < n) {
    float4 x = *(const float4*)(src + i);
    dst[i]     = (_Float16)x.x;
    dst[i + 1] = (_Float16)x.y;
    dst[i + 2] = (_Float16)x.z;
    dst[i + 3] = (_Float16)x.w;
  }
}

// ---------------------------------------------------------------------------
// Projection GEMM: Out[b,h,s,d] = f16( X[b*S+s, :] @ W[:, h*64+d] )
// X: [4096,1024] f16 row-major, W: [1024,1024] f16 row-major ([in,out]).
// ---------------------------------------------------------------------------
__global__ void __launch_bounds__(256)
proj_gemm(const _Float16* __restrict__ X, const _Float16* __restrict__ W,
          _Float16* __restrict__ Out) {
  __shared__ __align__(16) _Float16 As[BM * LDT];
  __shared__ __align__(16) _Float16 Bs[BN * LDT];
  const int tid = threadIdx.x, lane = tid & 31, wave = tid >> 5;
  const int gm0 = blockIdx.y * BM;
  const int gn0 = blockIdx.x * BN;

  const int ar = tid >> 1, ac = (tid & 1) * 16;      // A-tile copy coords
  const int kr = tid >> 3, ns = (tid & 7) * 8;       // B-tile copy coords

  v8f acc[4] = {};
  for (int k0 = 0; k0 < D_MODEL; k0 += BK) {
    // A tile: 128 rows x 32 halves -> async direct-to-LDS
    copy16_g2l(X + (size_t)(gm0 + ar) * D_MODEL + k0 + ac, As + ar * LDT + ac);
    copy16_g2l(X + (size_t)(gm0 + ar) * D_MODEL + k0 + ac + 8, As + ar * LDT + ac + 8);
    if (k0 + BK < D_MODEL)
      __builtin_prefetch(X + (size_t)(gm0 + ar) * D_MODEL + k0 + BK + ac, 0, 1);
    {  // B tile: W[k0..+32][gn0..+64] scattered transposed -> Bs[n][k]
      FragH tmp;
      *(uint4*)tmp.h = *(const uint4*)(W + (size_t)(k0 + kr) * D_MODEL + gn0 + ns);
#pragma unroll
      for (int j = 0; j < 8; ++j) Bs[(ns + j) * LDT + kr] = tmp.h[j];
    }
    async_wait0();
    __syncthreads();
    v16h a = lds_frag(As + wave * 16 * LDT, lane);
#pragma unroll
    for (int nt = 0; nt < 4; ++nt) {
      v16h b = lds_frag(Bs + nt * 16 * LDT, lane);
      acc[nt] = wmma_step(a, b, acc[nt]);
    }
    __syncthreads();
  }
  const int col = lane & 15, roff = (lane >> 4) * 8;
#pragma unroll
  for (int nt = 0; nt < 4; ++nt) {
    FragF cf; cf.v = acc[nt];
#pragma unroll
    for (int i = 0; i < 8; ++i) {
      int m = gm0 + wave * 16 + roff + i;
      int n = gn0 + nt * 16 + col;
      int b = m >> 11, s = m & (S_LEN - 1);
      int h = n >> 6,  d = n & (DK - 1);
      Out[((size_t)(b * NH + h) * S_LEN + s) * DK + d] = (_Float16)cf.f[i];
    }
  }
}

// ---------------------------------------------------------------------------
// Scores: Attn[b,h,sq,sk] = (Q[sq,:] . K[sk,:]) / sqrt(64)
// Qh/Kh: [b,h,s,64] f16. B tile = K rows directly (already [n][k] layout),
// so BOTH tiles are contiguous copies -> both go async direct-to-LDS.
// ---------------------------------------------------------------------------
__global__ void __launch_bounds__(256)
attn_scores(const _Float16* __restrict__ Qh, const _Float16* __restrict__ Kh,
            float* __restrict__ Attn) {
  __shared__ __align__(16) _Float16 As[BM * LDT];
  __shared__ __align__(16) _Float16 Bs[BN * LDT];
  const int tid = threadIdx.x, lane = tid & 31, wave = tid >> 5;
  const int z   = blockIdx.z;            // b*NH + h
  const int gm0 = blockIdx.y * BM;       // query rows
  const int gn0 = blockIdx.x * BN;       // key cols
  const _Float16* Q = Qh + (size_t)z * S_LEN * DK;
  const _Float16* K = Kh + (size_t)z * S_LEN * DK;

  const int ar = tid >> 1, ac = (tid & 1) * 16;
  const int br = tid >> 2, bc = (tid & 3) * 8;

  v8f acc[4] = {};
  for (int k0 = 0; k0 < DK; k0 += BK) {
    copy16_g2l(Q + (size_t)(gm0 + ar) * DK + k0 + ac,     As + ar * LDT + ac);
    copy16_g2l(Q + (size_t)(gm0 + ar) * DK + k0 + ac + 8, As + ar * LDT + ac + 8);
    copy16_g2l(K + (size_t)(gn0 + br) * DK + k0 + bc,     Bs + br * LDT + bc);
    async_wait0();
    __syncthreads();
    v16h a = lds_frag(As + wave * 16 * LDT, lane);
#pragma unroll
    for (int nt = 0; nt < 4; ++nt) {
      v16h b = lds_frag(Bs + nt * 16 * LDT, lane);
      acc[nt] = wmma_step(a, b, acc[nt]);
    }
    __syncthreads();
  }
  const float scale = 0.125f;  // 1/sqrt(DK)
  const int col = lane & 15, roff = (lane >> 4) * 8;
#pragma unroll
  for (int nt = 0; nt < 4; ++nt) {
    FragF cf; cf.v = acc[nt];
#pragma unroll
    for (int i = 0; i < 8; ++i) {
      int sq = gm0 + wave * 16 + roff + i;
      int sk = gn0 + nt * 16 + col;
      Attn[((size_t)z * S_LEN + sq) * S_LEN + sk] = cf.f[i] * scale;
    }
  }
}

// ---------------------------------------------------------------------------
// Row softmax over 2048 columns, in place. One block per row; row lives in
// registers (8 floats/thread), LDS tree reductions for max and sum.
// ---------------------------------------------------------------------------
__global__ void __launch_bounds__(256)
row_softmax(float* __restrict__ Attn) {
  __shared__ float red[256];
  float* p = Attn + (size_t)blockIdx.x * S_LEN;
  const int tid = threadIdx.x;
  float x[8];
#pragma unroll
  for (int i = 0; i < 8; ++i) x[i] = p[tid + i * 256];
  float m = x[0];
#pragma unroll
  for (int i = 1; i < 8; ++i) m = fmaxf(m, x[i]);
  red[tid] = m;
  __syncthreads();
  for (int s = 128; s > 0; s >>= 1) {
    if (tid < s) red[tid] = fmaxf(red[tid], red[tid + s]);
    __syncthreads();
  }
  m = red[0];
  __syncthreads();
  float sum = 0.f;
#pragma unroll
  for (int i = 0; i < 8; ++i) { x[i] = __expf(x[i] - m); sum += x[i]; }
  red[tid] = sum;
  __syncthreads();
  for (int s = 128; s > 0; s >>= 1) {
    if (tid < s) red[tid] += red[tid + s];
    __syncthreads();
  }
  float inv = 1.0f / red[0];
#pragma unroll
  for (int i = 0; i < 8; ++i) p[tid + i * 256] = x[i] * inv;
}

// ---------------------------------------------------------------------------
// PV: Ctx[b, sq, h*64+d] = f16( sum_k P[b,h,sq,k] * V[b,h,k,d] )
// P fp32 converted to f16 while staging into LDS (VGPR path required);
// V transposed-scatter into [d][k] layout.
// ---------------------------------------------------------------------------
__global__ void __launch_bounds__(256)
attn_pv(const float* __restrict__ Attn, const _Float16* __restrict__ Vh,
        _Float16* __restrict__ Ctx) {
  __shared__ __align__(16) _Float16 As[BM * LDT];
  __shared__ __align__(16) _Float16 Bs[BN * LDT];
  const int tid = threadIdx.x, lane = tid & 31, wave = tid >> 5;
  const int z   = blockIdx.z;
  const int gm0 = blockIdx.y * BM;
  const int b = z >> 4, h = z & 15;
  const float*    P = Attn + (size_t)z * S_LEN * S_LEN;
  const _Float16* V = Vh   + (size_t)z * S_LEN * DK;

  const int ar = tid >> 1, ac = (tid & 1) * 16;
  const int kr = tid >> 3, ns = (tid & 7) * 8;

  v8f acc[4] = {};
  for (int k0 = 0; k0 < S_LEN; k0 += BK) {
    {  // A: P rows fp32 -> f16
      const float4* src = (const float4*)(P + (size_t)(gm0 + ar) * S_LEN + k0 + ac);
      _Float16* dst = As + ar * LDT + ac;
#pragma unroll
      for (int j = 0; j < 4; ++j) {
        float4 f = src[j];
        dst[4*j]     = (_Float16)f.x;
        dst[4*j + 1] = (_Float16)f.y;
        dst[4*j + 2] = (_Float16)f.z;
        dst[4*j + 3] = (_Float16)f.w;
      }
      if (k0 + BK < S_LEN)
        __builtin_prefetch(P + (size_t)(gm0 + ar) * S_LEN + k0 + BK + ac, 0, 1);
    }
    {  // B: V[k0..+32][0..64] scattered transposed -> Bs[d][k]
      FragH tmp;
      *(uint4*)tmp.h = *(const uint4*)(V + (size_t)(k0 + kr) * DK + ns);
#pragma unroll
      for (int j = 0; j < 8; ++j) Bs[(ns + j) * LDT + kr] = tmp.h[j];
    }
    __syncthreads();
    v16h a = lds_frag(As + wave * 16 * LDT, lane);
#pragma unroll
    for (int nt = 0; nt < 4; ++nt) {
      v16h bfr = lds_frag(Bs + nt * 16 * LDT, lane);
      acc[nt] = wmma_step(a, bfr, acc[nt]);
    }
    __syncthreads();
  }
  const int col = lane & 15, roff = (lane >> 4) * 8;
#pragma unroll
  for (int nt = 0; nt < 4; ++nt) {
    FragF cf; cf.v = acc[nt];
#pragma unroll
    for (int i = 0; i < 8; ++i) {
      int sq = gm0 + wave * 16 + roff + i;
      int d  = nt * 16 + col;
      Ctx[((size_t)b * S_LEN + sq) * D_MODEL + h * DK + d] = (_Float16)cf.f[i];
    }
  }
}

// ---------------------------------------------------------------------------
// FC + residual: Out[m,n] = Ctx[m,:] @ Wfc[:,n] + Resid[m,n]   (fp32 out)
// ---------------------------------------------------------------------------
__global__ void __launch_bounds__(256)
fc_residual(const _Float16* __restrict__ Ctx, const _Float16* __restrict__ Wfc,
            const float* __restrict__ Resid, float* __restrict__ Out) {
  __shared__ __align__(16) _Float16 As[BM * LDT];
  __shared__ __align__(16) _Float16 Bs[BN * LDT];
  const int tid = threadIdx.x, lane = tid & 31, wave = tid >> 5;
  const int gm0 = blockIdx.y * BM;
  const int gn0 = blockIdx.x * BN;

  const int ar = tid >> 1, ac = (tid & 1) * 16;
  const int kr = tid >> 3, ns = (tid & 7) * 8;

  v8f acc[4] = {};
  for (int k0 = 0; k0 < D_MODEL; k0 += BK) {
    copy16_g2l(Ctx + (size_t)(gm0 + ar) * D_MODEL + k0 + ac,     As + ar * LDT + ac);
    copy16_g2l(Ctx + (size_t)(gm0 + ar) * D_MODEL + k0 + ac + 8, As + ar * LDT + ac + 8);
    if (k0 + BK < D_MODEL)
      __builtin_prefetch(Ctx + (size_t)(gm0 + ar) * D_MODEL + k0 + BK + ac, 0, 1);
    {
      FragH tmp;
      *(uint4*)tmp.h = *(const uint4*)(Wfc + (size_t)(k0 + kr) * D_MODEL + gn0 + ns);
#pragma unroll
      for (int j = 0; j < 8; ++j) Bs[(ns + j) * LDT + kr] = tmp.h[j];
    }
    async_wait0();
    __syncthreads();
    v16h a = lds_frag(As + wave * 16 * LDT, lane);
#pragma unroll
    for (int nt = 0; nt < 4; ++nt) {
      v16h b = lds_frag(Bs + nt * 16 * LDT, lane);
      acc[nt] = wmma_step(a, b, acc[nt]);
    }
    __syncthreads();
  }
  const int col = lane & 15, roff = (lane >> 4) * 8;
#pragma unroll
  for (int nt = 0; nt < 4; ++nt) {
    FragF cf; cf.v = acc[nt];
#pragma unroll
    for (int i = 0; i < 8; ++i) {
      int m = gm0 + wave * 16 + roff + i;
      int n = gn0 + nt * 16 + col;
      size_t idx = (size_t)m * D_MODEL + n;
      Out[idx] = cf.f[i] + Resid[idx];
    }
  }
}

// ---------------------------------------------------------------------------
// LayerNorm over D=1024, in place on Out. One block per row.
// ---------------------------------------------------------------------------
__global__ void __launch_bounds__(256)
layernorm(float* __restrict__ Out, const float* __restrict__ gamma,
          const float* __restrict__ beta) {
  __shared__ float red[256];
  float* p = Out + (size_t)blockIdx.x * D_MODEL;
  const int tid = threadIdx.x;
  float x[4];
  float s = 0.f;
#pragma unroll
  for (int i = 0; i < 4; ++i) { x[i] = p[tid + i * 256]; s += x[i]; }
  red[tid] = s;
  __syncthreads();
  for (int t = 128; t > 0; t >>= 1) {
    if (tid < t) red[tid] += red[tid + t];
    __syncthreads();
  }
  float mu = red[0] * (1.0f / D_MODEL);
  __syncthreads();
  float vs = 0.f;
#pragma unroll
  for (int i = 0; i < 4; ++i) { float d = x[i] - mu; vs += d * d; }
  red[tid] = vs;
  __syncthreads();
  for (int t = 128; t > 0; t >>= 1) {
    if (tid < t) red[tid] += red[tid + t];
    __syncthreads();
  }
  float inv = rsqrtf(red[0] * (1.0f / D_MODEL) + 1e-6f);
#pragma unroll
  for (int i = 0; i < 4; ++i) {
    int c = tid + i * 256;
    p[c] = (x[i] - mu) * inv * gamma[c] + beta[c];
  }
}

// ---------------------------------------------------------------------------
extern "C" void kernel_launch(void* const* d_in, const int* in_sizes, int n_in,
                              void* d_out, int out_size, void* d_ws, size_t ws_size,
                              hipStream_t stream) {
  (void)in_sizes; (void)n_in; (void)out_size; (void)ws_size;
  const float* q     = (const float*)d_in[0];
  const float* k     = (const float*)d_in[1];
  const float* v     = (const float*)d_in[2];
  const float* w_qs  = (const float*)d_in[3];
  const float* w_ks  = (const float*)d_in[4];
  const float* w_vs  = (const float*)d_in[5];
  const float* w_fc  = (const float*)d_in[6];
  const float* gamma = (const float*)d_in[7];
  const float* beta  = (const float*)d_in[8];

  float* out  = (float*)d_out;                                  // [B,S,D]
  float* attn = out + (size_t)B_BATCH * S_LEN * D_MODEL;        // [B,H,S,S]

  const size_t NQ = (size_t)B_BATCH * S_LEN * D_MODEL;  // 4,194,304
  const size_t NW = (size_t)D_MODEL * D_MODEL;          // 1,048,576
  _Float16* ws    = (_Float16*)d_ws;
  _Float16* q16   = ws;  ws += NQ;
  _Float16* k16   = ws;  ws += NQ;
  _Float16* v16   = ws;  ws += NQ;
  _Float16* wq16  = ws;  ws += NW;
  _Float16* wk16  = ws;  ws += NW;
  _Float16* wv16  = ws;  ws += NW;
  _Float16* wfc16 = ws;  ws += NW;
  _Float16* qh16  = ws;  ws += NQ;
  _Float16* kh16  = ws;  ws += NQ;
  _Float16* vh16  = ws;  ws += NQ;
  _Float16* ctx16 = ws;  ws += NQ;

  // 1) fp32 -> f16 staging
  cvt_f32_f16<<<(int)(NQ / 1024), 256, 0, stream>>>(q, q16, (int)NQ);
  cvt_f32_f16<<<(int)(NQ / 1024), 256, 0, stream>>>(k, k16, (int)NQ);
  cvt_f32_f16<<<(int)(NQ / 1024), 256, 0, stream>>>(v, v16, (int)NQ);
  cvt_f32_f16<<<(int)(NW / 1024), 256, 0, stream>>>(w_qs, wq16, (int)NW);
  cvt_f32_f16<<<(int)(NW / 1024), 256, 0, stream>>>(w_ks, wk16, (int)NW);
  cvt_f32_f16<<<(int)(NW / 1024), 256, 0, stream>>>(w_vs, wv16, (int)NW);
  cvt_f32_f16<<<(int)(NW / 1024), 256, 0, stream>>>(w_fc, wfc16, (int)NW);

  // 2) head projections (WMMA)
  dim3 gproj(D_MODEL / BN, (B_BATCH * S_LEN) / BM);
  proj_gemm<<<gproj, 256, 0, stream>>>(q16, wq16, qh16);
  proj_gemm<<<gproj, 256, 0, stream>>>(k16, wk16, kh16);
  proj_gemm<<<gproj, 256, 0, stream>>>(v16, wv16, vh16);

  // 3) scaled QK^T logits (WMMA) -> attn region of d_out
  dim3 gsc(S_LEN / BN, S_LEN / BM, B_BATCH * NH);
  attn_scores<<<gsc, 256, 0, stream>>>(qh16, kh16, attn);

  // 4) softmax in place
  row_softmax<<<B_BATCH * NH * S_LEN, 256, 0, stream>>>(attn);

  // 5) P @ V (WMMA) -> context
  dim3 gpv(1, S_LEN / BM, B_BATCH * NH);
  attn_pv<<<gpv, 256, 0, stream>>>(attn, vh16, ctx16);

  // 6) FC + residual (WMMA), then LayerNorm
  dim3 gfc(D_MODEL / BN, (B_BATCH * S_LEN) / BM);
  fc_residual<<<gfc, 256, 0, stream>>>(ctx16, wfc16, q, out);
  layernorm<<<B_BATCH * S_LEN, 256, 0, stream>>>(out, gamma, beta);
}
